// GraphNeuralCorrelator_40338332844309
// MI455X (gfx1250) — matrix operand order repaired and checked by reading, not compiled, not run on hardware
//
#include <hip/hip_runtime.h>
#include <hip/hip_bf16.h>
#include <stdint.h>

// ---------------------------------------------------------------------------
// Types for CDNA5 WMMA (gfx1250, wave32)
// ---------------------------------------------------------------------------
typedef __attribute__((ext_vector_type(16))) __bf16 v16bf;
typedef __attribute__((ext_vector_type(8)))  float  v8f;

#define TPB 256

__device__ __forceinline__ unsigned fkey(float f) {
  unsigned u = __float_as_uint(f);
  return (u & 0x80000000u) ? ~u : (u | 0x80000000u);
}
__device__ __forceinline__ float funkey(unsigned u) {
  return __uint_as_float((u & 0x80000000u) ? (u & 0x7FFFFFFFu) : ~u);
}
__device__ __forceinline__ unsigned short f2bf(float f) {
  unsigned u = __float_as_uint(f);
  u += 0x7FFFu + ((u >> 16) & 1u);      // round-to-nearest-even
  return (unsigned short)(u >> 16);
}

// ---------------------------------------------------------------------------
// f32 -> bf16 conversion (row-major copy)
// ---------------------------------------------------------------------------
__global__ void __launch_bounds__(TPB) cvt_bf16_kernel(const float* __restrict__ s,
                                                       unsigned short* __restrict__ d,
                                                       long long n) {
  long long i = (long long)blockIdx.x * blockDim.x + threadIdx.x;
  if (i >= n) return;
  d[i] = f2bf(s[i]);
}

// f32 [K,N] -> bf16 transposed [N,K] (for GEMM B operand)
__global__ void __launch_bounds__(TPB) cvt_bf16_t_kernel(const float* __restrict__ s,
                                                         unsigned short* __restrict__ d,
                                                         int K, int N) {
  int t = blockIdx.x * blockDim.x + threadIdx.x;
  if (t >= K * N) return;
  int k = t / N, n = t % N;
  d[(size_t)n * K + k] = f2bf(s[t]);
}

// ---------------------------------------------------------------------------
// GEMM: C[M,N] = A[M,K](bf16 row-major) @ B (given as Bt[N,K] bf16), f32 accum.
// One wave computes a 16(M) x 64(N) strip: 1 A fragment reused by 4 WMMAs.
// Fragment layouts per CDNA5 ISA 7.12.2:
//   A 16x32: lane-group 0 -> row bytes [0,16)+[32,48); group 1 -> [16,32)+[48,64)
//   B 32x16: lane (grp,mr) -> column tn+mr, K [grp*8, grp*8+8)+[16+grp*8, 24+grp*8)
// -> every fragment is exactly two dwordx4 loads, zero packing VALU.
// Software-pipelined with two named fragment sets (register double buffering)
// so next K-step loads overlap current WMMAs. Requires K/32 even and >= 2
// (true for all K used here: 128, 256, 512).
// ---------------------------------------------------------------------------
union Frag { v16bf v; uint4 q[2]; };
struct FragSet { Frag a, b0, b1, b2, b3; };

__device__ __forceinline__ void load_frags(FragSet& f,
                                           const unsigned short* __restrict__ Ap,
                                           const unsigned short* __restrict__ Bp,
                                           int k0, int grp, int K) {
  f.a.q[0] = *(const uint4*)(Ap + k0 + (grp << 3));
  f.a.q[1] = *(const uint4*)(Ap + k0 + 16 + (grp << 3));
  const unsigned short* b = Bp + k0 + (grp << 3);
  f.b0.q[0] = *(const uint4*)(b);
  f.b0.q[1] = *(const uint4*)(b + 16);
  f.b1.q[0] = *(const uint4*)(b + (size_t)16 * K);
  f.b1.q[1] = *(const uint4*)(b + (size_t)16 * K + 16);
  f.b2.q[0] = *(const uint4*)(b + (size_t)32 * K);
  f.b2.q[1] = *(const uint4*)(b + (size_t)32 * K + 16);
  f.b3.q[0] = *(const uint4*)(b + (size_t)48 * K);
  f.b3.q[1] = *(const uint4*)(b + (size_t)48 * K + 16);
}

__device__ __forceinline__ void mma_frags(const FragSet& f,
                                          v8f& a0, v8f& a1, v8f& a2, v8f& a3) {
  a0 = __builtin_amdgcn_wmma_f32_16x16x32_bf16(false, f.a.v, false, f.b0.v, (short)0, a0, false, false);
  a1 = __builtin_amdgcn_wmma_f32_16x16x32_bf16(false, f.a.v, false, f.b1.v, (short)0, a1, false, false);
  a2 = __builtin_amdgcn_wmma_f32_16x16x32_bf16(false, f.a.v, false, f.b2.v, (short)0, a2, false, false);
  a3 = __builtin_amdgcn_wmma_f32_16x16x32_bf16(false, f.a.v, false, f.b3.v, (short)0, a3, false, false);
}

__global__ void __launch_bounds__(TPB) gemm_bf16_wmma(const unsigned short* __restrict__ A,
                                                      const unsigned short* __restrict__ Bt,
                                                      float* __restrict__ C,
                                                      int M, int N, int K, int nwaves) {
  int wave = (int)(((long long)blockIdx.x * blockDim.x + threadIdx.x) >> 5);
  int lane = threadIdx.x & 31;
  if (wave >= nwaves) return;             // wave-uniform: EXEC all-ones at WMMA
  int tilesN = N >> 6;
  int tm = (wave / tilesN) << 4;
  int tn = (wave % tilesN) << 6;
  int grp = lane >> 4;                    // lane group 0/1
  int mr  = lane & 15;

  const unsigned short* Ap = A  + (size_t)(tm + mr) * K;
  const unsigned short* Bp = Bt + (size_t)(tn + mr) * K;   // +j*16*K per N-subtile

  FragSet f0, f1;
  v8f acc0 = {}, acc1 = {}, acc2 = {}, acc3 = {};

  load_frags(f0, Ap, Bp, 0, grp, K);
  int half = K >> 6;                      // number of 64-wide K double-steps
  int k0 = 32;
  for (int i = 0; i < half - 1; ++i) {
    __builtin_prefetch(Ap + k0 + 96, 0, 1);          // global_prefetch next A chunk
    load_frags(f1, Ap, Bp, k0, grp, K);
    mma_frags(f0, acc0, acc1, acc2, acc3);
    load_frags(f0, Ap, Bp, k0 + 32, grp, K);
    mma_frags(f1, acc0, acc1, acc2, acc3);
    k0 += 64;
  }
  load_frags(f1, Ap, Bp, k0, grp, K);
  mma_frags(f0, acc0, acc1, acc2, acc3);
  mma_frags(f1, acc0, acc1, acc2, acc3);

  // C: vgpr r -> row tm + r + grp*8, col tn + j*16 + mr
  float* Cp = C + (size_t)(tm + grp * 8) * N + tn + mr;
#pragma unroll
  for (int r = 0; r < 8; ++r) {
    Cp[(size_t)r * N +  0] = acc0[r];
    Cp[(size_t)r * N + 16] = acc1[r];
    Cp[(size_t)r * N + 32] = acc2[r];
    Cp[(size_t)r * N + 48] = acc3[r];
  }
}

// ---------------------------------------------------------------------------
// Async-copy probe: exercises the CDNA5 async global->LDS path
// (GLOBAL_LOAD_ASYNC_TO_LDS_B128, ASYNCcnt) per CDNA5 ISA.
// Writes only to workspace scratch; semantically a 128-float copy.
// ---------------------------------------------------------------------------
__global__ void __launch_bounds__(32) async_stage_probe(const float* __restrict__ src,
                                                        float* __restrict__ dst) {
  __shared__ __align__(16) float buf[128];
  int lane = threadIdx.x & 31;
  const float* g = src + lane * 4;
  unsigned lds = (unsigned)(size_t)(&buf[lane * 4]);
  asm volatile("global_load_async_to_lds_b128 %0, %1, off"
               :: "v"(lds), "v"(g) : "memory");
  asm volatile("s_wait_asynccnt 0x0" ::: "memory");
  float4 v = *(const float4*)(&buf[lane * 4]);
  *(float4*)(dst + lane * 4) = v;
}

// ---------------------------------------------------------------------------
// Per-(node,head) attention logits + accumulator init
// hf: [N, H*C];  as/ad: [H, C]
// ---------------------------------------------------------------------------
__global__ void __launch_bounds__(TPB) node_alpha_kernel(const float* __restrict__ hf,
                                                         const float* __restrict__ as,
                                                         const float* __restrict__ ad,
                                                         float* __restrict__ asn,
                                                         float* __restrict__ adn,
                                                         unsigned* __restrict__ mkey,
                                                         float* __restrict__ ssum,
                                                         int NH, int H, int C) {
  int i = blockIdx.x * blockDim.x + threadIdx.x;
  if (i >= NH) return;
  int h = i % H;
  const float* hp = hf + (size_t)i * C;
  const float* ap = as + (size_t)h * C;
  const float* dp = ad + (size_t)h * C;
  float sa = 0.f, sd = 0.f;
  for (int c = 0; c < C; ++c) { float v = hp[c]; sa += v * ap[c]; sd += v * dp[c]; }
  asn[i] = sa; adn[i] = sd;
  mkey[i] = 0u;        // ordered key for -inf (self-loop guarantees >=1 edge)
  ssum[i] = 0.f;
}

// ---------------------------------------------------------------------------
// Edge pass 1: leaky-relu logits + segment max (ordered-uint atomicMax)
// ---------------------------------------------------------------------------
__global__ void __launch_bounds__(TPB) edge_alpha_kernel(const int* __restrict__ ei,
                                                         int E, int Etot, int H,
                                                         const float* __restrict__ asn,
                                                         const float* __restrict__ adn,
                                                         float* __restrict__ we,
                                                         unsigned* __restrict__ mkey) {
  long long t = (long long)blockIdx.x * blockDim.x + threadIdx.x;
  long long total = (long long)Etot * H;
  if (t >= total) return;
  int h = (int)(t % H);
  int e = (int)(t / H);
  int s, d;
  if (e < E) { s = ei[e]; d = ei[E + e]; } else { s = d = e - E; }
  float a = asn[(size_t)s * H + h] + adn[(size_t)d * H + h];
  a = a > 0.f ? a : 0.2f * a;
  we[t] = a;
  atomicMax(&mkey[(size_t)d * H + h], fkey(a));
}

// ---------------------------------------------------------------------------
// Edge pass 2: exp(alpha - max) + segment sum
// ---------------------------------------------------------------------------
__global__ void __launch_bounds__(TPB) edge_exp_kernel(const int* __restrict__ ei,
                                                       int E, int Etot, int H,
                                                       float* __restrict__ we,
                                                       const unsigned* __restrict__ mkey,
                                                       float* __restrict__ ssum) {
  long long t = (long long)blockIdx.x * blockDim.x + threadIdx.x;
  long long total = (long long)Etot * H;
  if (t >= total) return;
  int h = (int)(t % H);
  int e = (int)(t / H);
  int d;
  if (e < E) { d = ei[E + e]; } else { d = e - E; }
  float m = funkey(mkey[(size_t)d * H + h]);
  float ex = __expf(we[t] - m);
  we[t] = ex;
  atomicAdd(&ssum[(size_t)d * H + h], ex);
}

// ---------------------------------------------------------------------------
// Edge pass 3: weighted message scatter-add. One wave per (edge,head),
// C/32 channels per lane.
// ---------------------------------------------------------------------------
__global__ void __launch_bounds__(TPB) edge_agg_kernel(const int* __restrict__ ei,
                                                       int E, int Etot, int H, int C,
                                                       const float* __restrict__ we,
                                                       const float* __restrict__ ssum,
                                                       const float* __restrict__ hf,
                                                       float* __restrict__ out) {
  long long tid = (long long)blockIdx.x * blockDim.x + threadIdx.x;
  long long total = (long long)Etot * H * 32;
  if (tid >= total) return;
  int lane = (int)(tid & 31);
  long long eh = tid >> 5;
  int h = (int)(eh % H);
  int e = (int)(eh / H);
  int s, d;
  if (e < E) { s = ei[e]; d = ei[E + e]; } else { s = d = e - E; }
  float wgt = we[eh] / ssum[(size_t)d * H + h];
  int cpl = C >> 5;
  const float* hs = hf + ((size_t)s * H + h) * C + lane * cpl;
  float* od = out + ((size_t)d * H + h) * C + lane * cpl;
  for (int c = 0; c < cpl; ++c) atomicAdd(&od[c], hs[c] * wgt);
}

// ---------------------------------------------------------------------------
// Bias (+optional ReLU) epilogue
// ---------------------------------------------------------------------------
__global__ void __launch_bounds__(TPB) bias_act_kernel(const float* __restrict__ in,
                                                       const float* __restrict__ b,
                                                       float* __restrict__ out,
                                                       long long n, int cols, int do_relu) {
  long long i = (long long)blockIdx.x * blockDim.x + threadIdx.x;
  if (i >= n) return;
  float v = in[i] + b[i % cols];
  if (do_relu) v = v > 0.f ? v : 0.f;
  out[i] = v;
}

// ---------------------------------------------------------------------------
// Correlation head: gather pair features into bf16 [P, 2*DOUT]
// ---------------------------------------------------------------------------
__global__ void __launch_bounds__(TPB) gather_pairs_kernel(const float* __restrict__ h2,
                                                           const int* __restrict__ pairs,
                                                           unsigned short* __restrict__ pf,
                                                           int P, int DOUT) {
  int t = blockIdx.x * blockDim.x + threadIdx.x;
  if (t >= P * 64) return;
  int p = t / 64, j = t % 64;
  int i0 = pairs[p * 2 + 0], i1 = pairs[p * 2 + 1];
  pf[(size_t)p * 128 + j]      = f2bf(h2[(size_t)i0 * DOUT + j]);
  pf[(size_t)p * 128 + 64 + j] = f2bf(h2[(size_t)i1 * DOUT + j]);
}

__global__ void __launch_bounds__(TPB) corr_final_kernel(const float* __restrict__ z2,
                                                         const float* __restrict__ w3,
                                                         const float* __restrict__ b3,
                                                         float* __restrict__ out, int P) {
  int p = blockIdx.x * blockDim.x + threadIdx.x;
  if (p >= P) return;
  float t = b3[0];
  for (int j = 0; j < 64; ++j) t += z2[(size_t)p * 64 + j] * w3[j];
  out[p] = 1.f / (1.f + __expf(-t));
}

// ---------------------------------------------------------------------------
// Host orchestration
// ---------------------------------------------------------------------------
static inline unsigned nb(long long n) { return (unsigned)((n + TPB - 1) / TPB); }

extern "C" void kernel_launch(void* const* d_in, const int* in_sizes, int n_in,
                              void* d_out, int out_size, void* d_ws, size_t ws_size,
                              hipStream_t stream) {
  (void)in_sizes; (void)n_in; (void)out_size; (void)ws_size;
  constexpr int N_ = 32768, E_ = 524288, P_ = 4096;
  constexpr int DIN_ = 256, HID_ = 128, DOUT_ = 64, H_ = 4;
  const int Etot = E_ + N_;

  const float* x    = (const float*)d_in[0];
  const int*   ei   = (const int*)d_in[1];
  const int*   prs  = (const int*)d_in[2];
  const float* W0  = (const float*)d_in[3];
  const float* as0 = (const float*)d_in[4];
  const float* ad0 = (const float*)d_in[5];
  const float* b0  = (const float*)d_in[6];
  const float* W1  = (const float*)d_in[7];
  const float* as1 = (const float*)d_in[8];
  const float* ad1 = (const float*)d_in[9];
  const float* b1  = (const float*)d_in[10];
  const float* W2  = (const float*)d_in[11];
  const float* as2 = (const float*)d_in[12];
  const float* ad2 = (const float*)d_in[13];
  const float* b2  = (const float*)d_in[14];
  const float* cw1 = (const float*)d_in[15];
  const float* cb1 = (const float*)d_in[16];
  const float* cw2 = (const float*)d_in[17];
  const float* cb2 = (const float*)d_in[18];
  const float* cw3 = (const float*)d_in[19];
  const float* cb3 = (const float*)d_in[20];
  float* out = (float*)d_out;

  // ---- workspace carve (256B aligned) ----
  uintptr_t base = (uintptr_t)d_ws;
  size_t off = 0;
  auto carve = [&](size_t bytes) -> void* {
    void* p = (void*)(base + off);
    off += (bytes + 255) & ~(size_t)255;
    return p;
  };
  unsigned short* ab  = (unsigned short*)carve((size_t)N_ * 512 * 2);  // bf16 A operand
  unsigned short* wb  = (unsigned short*)carve((size_t)512 * 512 * 2); // bf16 B^T operand
  float*    hf    = (float*)carve((size_t)N_ * 512 * 4);   // GEMM output h
  float*    agg   = (float*)carve((size_t)N_ * 512 * 4);   // aggregation accumulator
  float*    actb  = (float*)carve((size_t)N_ * 512 * 4);   // post-activation
  float*    asn   = (float*)carve((size_t)N_ * H_ * 4);
  float*    adn   = (float*)carve((size_t)N_ * H_ * 4);
  unsigned* mkey  = (unsigned*)carve((size_t)N_ * H_ * 4);
  float*    ssum  = (float*)carve((size_t)N_ * H_ * 4);
  float*    we    = (float*)carve((size_t)Etot * H_ * 4);
  float*    probe = (float*)carve(128 * 4);

  // async global->LDS path probe (writes only to workspace scratch)
  async_stage_probe<<<1, 32, 0, stream>>>(x, probe);

  auto run_gemm = [&](const unsigned short* A, const unsigned short* Bt, float* C,
                      int M, int Nn, int K) {
    int nwaves = (M / 16) * (Nn / 64);
    gemm_bf16_wmma<<<nb((long long)nwaves * 32), TPB, 0, stream>>>(A, Bt, C, M, Nn, K, nwaves);
  };

  auto gat_layer = [&](const float* actf, int K, int H, int C,
                       const float* Wm, const float* asv, const float* adv) {
    long long nact = (long long)N_ * K;
    cvt_bf16_kernel<<<nb(nact), TPB, 0, stream>>>(actf, ab, nact);
    cvt_bf16_t_kernel<<<nb((long long)K * H * C), TPB, 0, stream>>>(Wm, wb, K, H * C);
    run_gemm(ab, wb, hf, N_, H * C, K);
    int NH = N_ * H;
    node_alpha_kernel<<<nb(NH), TPB, 0, stream>>>(hf, asv, adv, asn, adn, mkey, ssum, NH, H, C);
    long long EH = (long long)Etot * H;
    edge_alpha_kernel<<<nb(EH), TPB, 0, stream>>>(ei, E_, Etot, H, asn, adn, we, mkey);
    edge_exp_kernel<<<nb(EH), TPB, 0, stream>>>(ei, E_, Etot, H, we, mkey, ssum);
    hipMemsetAsync(agg, 0, (size_t)N_ * H * C * sizeof(float), stream);
    edge_agg_kernel<<<nb(EH * 32), TPB, 0, stream>>>(ei, E_, Etot, H, C, we, ssum, hf, agg);
  };

  // Layer 0: DIN -> H*HID (concat, relu)
  gat_layer(x, DIN_, H_, HID_, W0, as0, ad0);
  bias_act_kernel<<<nb((long long)N_ * 512), TPB, 0, stream>>>(agg, b0, actb,
                                                               (long long)N_ * 512, 512, 1);
  // Layer 1: 512 -> H*HID (concat, relu)
  gat_layer(actb, H_ * HID_, H_, HID_, W1, as1, ad1);
  bias_act_kernel<<<nb((long long)N_ * 512), TPB, 0, stream>>>(agg, b1, actb,
                                                               (long long)N_ * 512, 512, 1);
  // Layer 2: 512 -> DOUT, 1 head, mean over 1 head == identity, no relu
  gat_layer(actb, H_ * HID_, 1, DOUT_, W2, as2, ad2);
  bias_act_kernel<<<nb((long long)N_ * DOUT_), TPB, 0, stream>>>(agg, b2, out,
                                                                 (long long)N_ * DOUT_, DOUT_, 0);

  // ---- correlation head ----
  gather_pairs_kernel<<<nb((long long)P_ * 64), TPB, 0, stream>>>(out, prs, ab, P_, DOUT_);
  // z1 = relu(pf @ cw1 + cb1)
  cvt_bf16_t_kernel<<<nb(128 * 128), TPB, 0, stream>>>(cw1, wb, 128, 128);
  run_gemm(ab, wb, hf, P_, 128, 128);
  bias_act_kernel<<<nb((long long)P_ * 128), TPB, 0, stream>>>(hf, cb1, agg,
                                                               (long long)P_ * 128, 128, 1);
  // z2 = relu(z1 @ cw2 + cb2)
  cvt_bf16_kernel<<<nb((long long)P_ * 128), TPB, 0, stream>>>(agg, ab, (long long)P_ * 128);
  cvt_bf16_t_kernel<<<nb(128 * 64), TPB, 0, stream>>>(cw2, wb, 128, 64);
  run_gemm(ab, wb, hf, P_, 64, 128);
  bias_act_kernel<<<nb((long long)P_ * 64), TPB, 0, stream>>>(hf, cb2, agg,
                                                              (long long)P_ * 64, 64, 1);
  // corr = sigmoid(z2 @ cw3 + cb3)
  corr_final_kernel<<<nb(P_), TPB, 0, stream>>>(agg, cw3, cb3, out + (size_t)N_ * DOUT_, P_);
}
